// TransformerEncoderLayer_48799418417542
// MI455X (gfx1250) — matrix operand order repaired and checked by reading, not compiled, not run on hardware
//
#include <hip/hip_runtime.h>

// ---------------------------------------------------------------------------
// Fused relative-attention transformer layer for MI455X (gfx1250, wave32).
// Dominant cost: one streaming pass over rel_features (134 MB -> ~5.8us at
// 23.3 TB/s). rel_emb [B,L,L,D] (1 GB) is never materialized: contract q with
// rel_w first (qw), then WMMA-contract rel_features x qw over C=64.
//   - f32 WMMA (16x16x4) for attention scores (softmax-precision path)
//   - bf16 WMMA (16x16x32) for the FFN GEMMs (f32 accumulate)
//   - rel_features tiles streamed via global_load_async_to_lds_b128 with
//     per-wave double buffering + s_wait_asynccnt (ASYNCcnt) synchronization
// ---------------------------------------------------------------------------

typedef __attribute__((ext_vector_type(2)))  float  v2f;
typedef __attribute__((ext_vector_type(8)))  float  v8f;
typedef __attribute__((ext_vector_type(16))) __bf16 v16bf;

constexpr int   B_   = 2;
constexpr int   L_   = 512;
constexpr int   D_   = 512;
constexpr int   H_   = 8;
constexpr int   DH_  = 64;
constexpr int   C_   = 64;
constexpr int   FF_  = 2048;
constexpr float SCALE_ = 0.125f;     // 1/sqrt(64)
constexpr float EPS_   = 1e-5f;

#define DEV __device__ __forceinline__

DEV __bf16 f2bf(float f) {
  union { float f; unsigned u; } x; x.f = f;
  unsigned u = x.u + 0x7FFFu + ((x.u >> 16) & 1u);   // round-nearest-even
  union { unsigned short s; __bf16 b; } y; y.s = (unsigned short)(u >> 16);
  return y.b;
}

// D = A(16x4) * B(4x16) + C, f32. Hardware WMMA confirmed by round-1 codegen;
// shuffle-emulation fallback retained as a safety net.
DEV v8f wmma_f32_k4(v2f a, v2f b, v8f c) {
#if __has_builtin(__builtin_amdgcn_wmma_f32_16x16x4_f32)
  return __builtin_amdgcn_wmma_f32_16x16x4_f32(false, a, false, b, (short)0, c,
                                               false, false);
#else
  int lane = (int)(threadIdx.x & 31u);
  int n = lane & 15;
  #pragma unroll
  for (int r = 0; r < 8; ++r) {
    int m = r + ((lane >> 4) << 3);
    float d = c[r];
    d += __shfl(a[0], m)      * __shfl(b[0], n);
    d += __shfl(a[1], m)      * __shfl(b[1], n);
    d += __shfl(a[0], m + 16) * __shfl(b[0], n + 16);
    d += __shfl(a[1], m + 16) * __shfl(b[1], n + 16);
    c[r] = d;
  }
  return c;
#endif
}

// Async-DMA one contiguous 4KB tile (16 rows x 64 f32) from global into LDS.
// Per-lane base covers lane*16B; the shared IOFFSET advances both the global
// and the LDS address (ISA 08_async_tensor 4.4), so 8 issues move 4KB.
// Each issue bumps ASYNCcnt by 1.
#define ASYNC_COPY_4K(ldsoff, gaddr)                                          \
  asm volatile(                                                               \
      "global_load_async_to_lds_b128 %0, %1, off\n\t"                         \
      "global_load_async_to_lds_b128 %0, %1, off offset:512\n\t"              \
      "global_load_async_to_lds_b128 %0, %1, off offset:1024\n\t"             \
      "global_load_async_to_lds_b128 %0, %1, off offset:1536\n\t"             \
      "global_load_async_to_lds_b128 %0, %1, off offset:2048\n\t"             \
      "global_load_async_to_lds_b128 %0, %1, off offset:2560\n\t"             \
      "global_load_async_to_lds_b128 %0, %1, off offset:3072\n\t"             \
      "global_load_async_to_lds_b128 %0, %1, off offset:3584"                 \
      ::"v"(ldsoff), "v"(gaddr)                                               \
      : "memory")

#define WAIT_ASYNC_LE8()  asm volatile("s_wait_asynccnt 0x8" ::: "memory")
#define WAIT_ASYNC_ZERO() asm volatile("s_wait_asynccnt 0x0" ::: "memory")

// --------------------------- K0: weights -> bf16 ---------------------------
__global__ __launch_bounds__(256) void k_cvt_w(const float* __restrict__ w1,
                                               const float* __restrict__ w2,
                                               __bf16* __restrict__ w1b,
                                               __bf16* __restrict__ w2b) {
  size_t idx = (size_t)blockIdx.x * 256 + threadIdx.x;   // FF_*D_ threads
  w1b[idx] = f2bf(w1[idx]);
  w2b[idx] = f2bf(w2[idx]);
}

// --------- K1: qw[b,i,h,c] = q . rel_w ; qb[b,i,h] = q . rel_b -------------
__global__ __launch_bounds__(256) void k_qproj(const float* __restrict__ src,
                                               const float* __restrict__ rel_w,
                                               const float* __restrict__ rel_b,
                                               float* __restrict__ qw,
                                               float* __restrict__ qb) {
  __shared__ float s[D_];
  int row = blockIdx.x;            // b*L + i
  int t = threadIdx.x;
  s[t]       = src[(size_t)row * D_ + t];
  s[t + 256] = src[(size_t)row * D_ + t + 256];
  __syncthreads();
  for (int p = t; p < H_ * C_; p += 256) {
    int h = p >> 6, c = p & 63;
    float a = 0.f;
    #pragma unroll 8
    for (int d = 0; d < DH_; ++d)
      a += s[h * DH_ + d] * rel_w[(h * DH_ + d) * C_ + c];
    qw[(size_t)row * (H_ * C_) + p] = a;
  }
  if (t < H_) {
    float a = 0.f;
    #pragma unroll 8
    for (int d = 0; d < DH_; ++d) a += s[t * DH_ + d] * rel_b[t * DH_ + d];
    qb[(size_t)row * H_ + t] = a;
  }
}

// ------- K2: base_scores[b,h] = (scale*Q_h) @ Q_h^T  (f32 WMMA GEMM) -------
__global__ __launch_bounds__(256) void k_base(const float* __restrict__ src,
                                              float* __restrict__ sbase) {
  int itile = blockIdx.x & 31;
  int h     = (blockIdx.x >> 5) & 7;
  int b     = blockIdx.x >> 8;
  int lane = threadIdx.x & 31, wave = threadIdx.x >> 5;
  int m = lane & 15, koff = (lane >> 4) * 2;

  const float* qrow = src + ((size_t)(b * L_ + itile * 16 + m)) * D_ + h * DH_;
  v2f afr[16];                                  // A frags held for 4 N-tiles
  #pragma unroll
  for (int s = 0; s < 16; ++s) {
    int cc = 4 * s + koff;
    afr[s] = v2f{qrow[cc] * SCALE_, qrow[cc + 1] * SCALE_};
  }
  for (int tix = 0; tix < 4; ++tix) {
    int jt = (wave * 4 + tix) * 16;
    int n = lane & 15;
    const float* krow = src + ((size_t)(b * L_ + jt + n)) * D_ + h * DH_ + koff;
    v8f c = {0.f, 0.f, 0.f, 0.f, 0.f, 0.f, 0.f, 0.f};
    #pragma unroll
    for (int s = 0; s < 16; ++s) {
      v2f bf = *(const v2f*)(krow + 4 * s);
      c = wmma_f32_k4(afr[s], bf, c);
    }
    #pragma unroll
    for (int r = 0; r < 8; ++r) {
      int irow = itile * 16 + r + ((lane >> 4) << 3);
      sbase[(((size_t)(b * H_ + h)) * L_ + irow) * L_ + jt + n] = c[r];
    }
  }
}

// --- K3: per-(b,i) fused rel-WMMA scores + softmax + attn@V + LN1 ----------
// Dynamic LDS layout (floats):
//   [0,16384)  per-wave double-buffered rel_features tiles (8 x 2 x 1024)
//   [16384,..) scores(4096) | qw(1024) | qi(512) | xrow(512) | reductions
__global__ __launch_bounds__(256) void k_attn(
    const float* __restrict__ src, const float* __restrict__ rf,
    const float* __restrict__ qw, const float* __restrict__ qb,
    const float* __restrict__ sbase, const float* __restrict__ g1,
    const float* __restrict__ be1, float* __restrict__ xbuf,
    __bf16* __restrict__ xbf) {
  extern __shared__ float smem[];
  float* stageF = smem;                      // 64 KB staging
  float* scL  = stageF + H_ * 2 * 16 * C_;   // 8*512
  float* qwL  = scL + H_ * L_;               // 16*64 (heads padded to 16)
  float* qiL  = qwL + 16 * C_;               // 512
  float* xrL  = qiL + D_;                    // 512
  float* redS = xrL + D_;                    // 8
  float* redQ = redS + H_;                   // 8
  float* stat = redQ + H_;                   // 2

  int b = blockIdx.x / L_, i = blockIdx.x % L_;
  int t = threadIdx.x, lane = t & 31, wave = t >> 5;
  size_t rowb = (size_t)(b * L_ + i);

  // Kick off the async DMA of this wave's first rel_features tile ASAP.
  const float* rfb = rf + rowb * (size_t)(L_ * C_);   // this i's [L,C] slab
  unsigned ldsA = (unsigned)(uintptr_t)(stageF + (size_t)wave * 2048) +
                  (unsigned)(lane * 16);
  unsigned ldsB = ldsA + 4096u;
  uint64_t gbase = (uint64_t)(uintptr_t)rfb + (uint64_t)(wave * 16384) +
                   (uint64_t)(lane * 16);
  ASYNC_COPY_4K(ldsA, gbase);

  qiL[t]       = src[rowb * D_ + t];
  qiL[t + 256] = src[rowb * D_ + t + 256];
  for (int p = t; p < 16 * C_; p += 256) {
    int r = p >> 6, c = p & 63;
    qwL[p] = (r < H_) ? qw[rowb * (H_ * C_) + r * C_ + c] : 0.f;
  }
  __syncthreads();

  int m = lane & 15, koff = (lane >> 4) * 2, n = lane & 15;
  v2f afr[16];
  #pragma unroll
  for (int s = 0; s < 16; ++s) {
    int cc = 4 * s + koff;
    afr[s] = v2f{qwL[m * C_ + cc], qwL[m * C_ + cc + 1]};
  }
  float qbr[8];
  #pragma unroll
  for (int r = 0; r < 8; ++r) qbr[r] = qb[rowb * H_ + r];

  for (int tix = 0; tix < 4; ++tix) {
    int jt = (wave * 4 + tix) * 16;
    v8f c;
    #pragma unroll
    for (int r = 0; r < 8; ++r) {     // C-init: base scores + qb (valid rows
      float v = 0.f;                  // live in lanes 0..15; pad rows -> 0)
      if (lane < 16)
        v = sbase[(((size_t)(b * H_ + r)) * L_ + i) * L_ + jt + n] + qbr[r];
      c[r] = v;
    }
    // Prefetch tile tix+1 into the other LDS buffer, then wait only for the
    // current tile (in-order ASYNCcnt completion: <=8 outstanding means the
    // previous batch of 8 b128 DMAs has landed).
    if (tix < 3) {
      ASYNC_COPY_4K((tix & 1) ? ldsA : ldsB, gbase + (uint64_t)((tix + 1) * 4096));
      WAIT_ASYNC_LE8();
    } else {
      WAIT_ASYNC_ZERO();
    }
    const float* st = stageF + (size_t)(wave * 2 + (tix & 1)) * 1024;
    const float* strow = st + n * C_ + koff;
    #pragma unroll
    for (int s = 0; s < 16; ++s) {    // K=64 in 16 x (K=4) WMMA steps
      v2f bf = *(const v2f*)(strow + 4 * s);
      c = wmma_f32_k4(afr[s], bf, c);
    }
    if (lane < 16) {
      #pragma unroll
      for (int r = 0; r < 8; ++r) scL[r * L_ + jt + lane] = c[r];
    }
  }
  __syncthreads();

  {  // wave == head: softmax over j, then attn @ V for this head
    int h = wave;
    float mx = -3.4e38f;
    #pragma unroll
    for (int u = 0; u < 16; ++u) mx = fmaxf(mx, scL[h * L_ + lane + 32 * u]);
    for (int o = 16; o > 0; o >>= 1) mx = fmaxf(mx, __shfl_xor(mx, o));
    float sum = 0.f;
    #pragma unroll
    for (int u = 0; u < 16; ++u) {
      int ix = h * L_ + lane + 32 * u;
      float e = __expf(scL[ix] - mx);
      scL[ix] = e;
      sum += e;
    }
    for (int o = 16; o > 0; o >>= 1) sum += __shfl_xor(sum, o);
    float rinv = 1.f / sum;
    #pragma unroll
    for (int u = 0; u < 16; ++u) scL[h * L_ + lane + 32 * u] *= rinv;

    int d0 = 2 * lane;
    const float* vsrc = src + (size_t)b * L_ * D_ + h * DH_ + d0;
    float a0 = 0.f, a1 = 0.f;
    for (int j = 0; j < L_; ++j) {
      float p = scL[h * L_ + j];
      v2f vv = *(const v2f*)(vsrc + (size_t)j * D_);
      a0 += p * vv[0];
      a1 += p * vv[1];
    }
    xrL[h * DH_ + d0]     = qiL[h * DH_ + d0]     + a0;
    xrL[h * DH_ + d0 + 1] = qiL[h * DH_ + d0 + 1] + a1;
  }
  __syncthreads();

  // LN1 across D=512
  float x0 = xrL[t], x1 = xrL[t + 256];
  float ps = x0 + x1, pq = x0 * x0 + x1 * x1;
  for (int o = 16; o > 0; o >>= 1) { ps += __shfl_xor(ps, o); pq += __shfl_xor(pq, o); }
  if (lane == 0) { redS[wave] = ps; redQ[wave] = pq; }
  __syncthreads();
  if (t == 0) {
    float S = 0.f, Q = 0.f;
    for (int w = 0; w < H_; ++w) { S += redS[w]; Q += redQ[w]; }
    float mu = S / (float)D_;
    stat[0] = mu;
    stat[1] = rsqrtf(Q / (float)D_ - mu * mu + EPS_);
  }
  __syncthreads();
  float mu = stat[0], rs = stat[1];
  #pragma unroll
  for (int e = 0; e < 2; ++e) {
    int d = t + e * 256;
    float xn = (xrL[d] - mu) * rs * g1[d] + be1[d];
    xbuf[rowb * D_ + d] = xn;
    xbf[rowb * D_ + d]  = f2bf(xn);
  }
}

// --------------- K4: FFN1 = leaky_relu(x @ w1^T + b1), bf16 WMMA -----------
__global__ __launch_bounds__(256) void k_ffn1(const __bf16* __restrict__ xb,
                                              const __bf16* __restrict__ w1b,
                                              const float* __restrict__ b1,
                                              __bf16* __restrict__ hbf) {
  int mtile = blockIdx.x >> 4, ngrp = blockIdx.x & 15;
  int lane = threadIdx.x & 31, wave = threadIdx.x >> 5;
  int ntile = ngrp * 8 + wave;
  int m = lane & 15, n = lane & 15, sk = lane >> 4;
  int row = mtile * 16 + m, col0 = ntile * 16;
  const __bf16* ar = xb  + (size_t)row * D_;
  const __bf16* br = w1b + (size_t)(col0 + n) * D_;
  v8f acc = {0.f, 0.f, 0.f, 0.f, 0.f, 0.f, 0.f, 0.f};
  for (int k0 = 0; k0 < D_; k0 += 32) {
    if (k0 + 32 < D_) {
      __builtin_prefetch(ar + k0 + 32, 0, 1);   // global_prefetch_b8
      __builtin_prefetch(br + k0 + 32, 0, 1);
    }
    v16bf a, bb;
    #pragma unroll
    for (int v = 0; v < 8; ++v) {
      int ka = k0 + ((v & 3) << 1) + ((v >> 2) << 4) + (sk << 3);  // A layout
      a[2 * v] = ar[ka];  a[2 * v + 1] = ar[ka + 1];
      int kb = k0 + ((v & 3) << 1) + ((v >> 2) << 4) + (sk << 3);  // B layout
      bb[2 * v] = br[kb]; bb[2 * v + 1] = br[kb + 1];
    }
    acc = __builtin_amdgcn_wmma_f32_16x16x32_bf16(false, a, false, bb,
                                                  (short)0, acc, false, false);
  }
  int col = col0 + n;
  float bias = b1[col];
  #pragma unroll
  for (int r = 0; r < 8; ++r) {
    int orow = mtile * 16 + r + (sk << 3);
    float v = acc[r] + bias;
    v = v > 0.f ? v : 0.01f * v;                 // leaky relu
    hbf[(size_t)orow * FF_ + col] = f2bf(v);
  }
}

// ------------- K5: FFN2 = h @ w2^T + b2 + x (residual), bf16 WMMA ----------
__global__ __launch_bounds__(256) void k_ffn2(const __bf16* __restrict__ hb,
                                              const __bf16* __restrict__ w2b,
                                              const float* __restrict__ b2,
                                              const float* __restrict__ xbuf,
                                              float* __restrict__ ybuf) {
  int mtile = blockIdx.x >> 2, ngrp = blockIdx.x & 3;
  int lane = threadIdx.x & 31, wave = threadIdx.x >> 5;
  int ntile = ngrp * 8 + wave;
  int m = lane & 15, n = lane & 15, sk = lane >> 4;
  int row = mtile * 16 + m, col0 = ntile * 16;
  const __bf16* ar = hb  + (size_t)row * FF_;
  const __bf16* br = w2b + (size_t)(col0 + n) * FF_;
  v8f acc = {0.f, 0.f, 0.f, 0.f, 0.f, 0.f, 0.f, 0.f};
  for (int k0 = 0; k0 < FF_; k0 += 32) {
    if (k0 + 32 < FF_) {
      __builtin_prefetch(ar + k0 + 32, 0, 1);
      __builtin_prefetch(br + k0 + 32, 0, 1);
    }
    v16bf a, bb;
    #pragma unroll
    for (int v = 0; v < 8; ++v) {
      int ka = k0 + ((v & 3) << 1) + ((v >> 2) << 4) + (sk << 3);
      a[2 * v] = ar[ka];  a[2 * v + 1] = ar[ka + 1];
      int kb = k0 + ((v & 3) << 1) + ((v >> 2) << 4) + (sk << 3);
      bb[2 * v] = br[kb]; bb[2 * v + 1] = br[kb + 1];
    }
    acc = __builtin_amdgcn_wmma_f32_16x16x32_bf16(false, a, false, bb,
                                                  (short)0, acc, false, false);
  }
  int col = col0 + n;
  float bias = b2[col];
  #pragma unroll
  for (int r = 0; r < 8; ++r) {
    int orow = mtile * 16 + r + (sk << 3);
    ybuf[(size_t)orow * D_ + col] =
        acc[r] + bias + xbuf[(size_t)orow * D_ + col];
  }
}

// ------------------------------- K6: LN2 -----------------------------------
__global__ __launch_bounds__(256) void k_ln2(const float* __restrict__ y,
                                             const float* __restrict__ g2,
                                             const float* __restrict__ be2,
                                             float* __restrict__ out) {
  __shared__ float redS[8], redQ[8], stat[2];
  int row = blockIdx.x, t = threadIdx.x, lane = t & 31, wave = t >> 5;
  float x0 = y[(size_t)row * D_ + t], x1 = y[(size_t)row * D_ + t + 256];
  float ps = x0 + x1, pq = x0 * x0 + x1 * x1;
  for (int o = 16; o > 0; o >>= 1) { ps += __shfl_xor(ps, o); pq += __shfl_xor(pq, o); }
  if (lane == 0) { redS[wave] = ps; redQ[wave] = pq; }
  __syncthreads();
  if (t == 0) {
    float S = 0.f, Q = 0.f;
    for (int w = 0; w < 8; ++w) { S += redS[w]; Q += redQ[w]; }
    float mu = S / (float)D_;
    stat[0] = mu;
    stat[1] = rsqrtf(Q / (float)D_ - mu * mu + EPS_);
  }
  __syncthreads();
  float mu = stat[0], rs = stat[1];
  out[(size_t)row * D_ + t]       = (x0 - mu) * rs * g2[t] + be2[t];
  out[(size_t)row * D_ + t + 256] = (x1 - mu) * rs * g2[t + 256] + be2[t + 256];
}

// ---------------------------------------------------------------------------
extern "C" void kernel_launch(void* const* d_in, const int* in_sizes, int n_in,
                              void* d_out, int out_size, void* d_ws,
                              size_t ws_size, hipStream_t stream) {
  const float* src   = (const float*)d_in[0];
  const float* rf    = (const float*)d_in[1];
  const float* rel_w = (const float*)d_in[2];
  const float* rel_b = (const float*)d_in[3];
  const float* w1    = (const float*)d_in[4];
  const float* b1    = (const float*)d_in[5];
  const float* w2    = (const float*)d_in[6];
  const float* b2    = (const float*)d_in[7];
  const float* g1    = (const float*)d_in[8];
  const float* be1   = (const float*)d_in[9];
  const float* g2    = (const float*)d_in[10];
  const float* be2   = (const float*)d_in[11];
  float* out = (float*)d_out;

  // workspace carve-out (~31.3 MB)
  char* ws = (char*)d_ws;
  size_t off = 0;
  auto carve = [&](size_t bytes) -> void* {
    void* p = ws + off;
    off = (off + bytes + 255) & ~(size_t)255;
    return p;
  };
  float*  qwp   = (float*) carve(sizeof(float)  * B_ * L_ * H_ * C_);   // 2 MB
  float*  qbp   = (float*) carve(sizeof(float)  * B_ * L_ * H_);        // 32 KB
  float*  sbase = (float*) carve(sizeof(float)  * B_ * H_ * L_ * L_);   // 16 MB
  float*  xbuf  = (float*) carve(sizeof(float)  * B_ * L_ * D_);        // 2 MB
  float*  ybuf  = (float*) carve(sizeof(float)  * B_ * L_ * D_);        // 2 MB
  __bf16* xbf   = (__bf16*)carve(sizeof(__bf16) * B_ * L_ * D_);        // 1 MB
  __bf16* w1bf  = (__bf16*)carve(sizeof(__bf16) * FF_ * D_);            // 2 MB
  __bf16* w2bf  = (__bf16*)carve(sizeof(__bf16) * D_ * FF_);            // 2 MB
  __bf16* hbf   = (__bf16*)carve(sizeof(__bf16) * B_ * L_ * FF_);       // 4 MB
  (void)ws_size; (void)in_sizes; (void)n_in; (void)out_size;

  // k_attn dynamic LDS: 64KB staging + 24.125KB working set
  constexpr int SMEM_ATTN =
      (int)((H_ * 2 * 16 * C_ + H_ * L_ + 16 * C_ + D_ + D_ + 32) *
            sizeof(float));                                   // 90,240 B
  hipFuncSetAttribute((const void*)k_attn,
                      hipFuncAttributeMaxDynamicSharedMemorySize, SMEM_ATTN);

  k_cvt_w<<<(FF_ * D_) / 256, 256, 0, stream>>>(w1, w2, w1bf, w2bf);
  k_qproj<<<B_ * L_, 256, 0, stream>>>(src, rel_w, rel_b, qwp, qbp);
  k_base <<<B_ * H_ * (L_ / 16), 256, 0, stream>>>(src, sbase);
  k_attn <<<B_ * L_, 256, SMEM_ATTN, stream>>>(src, rf, qwp, qbp, sbase, g1,
                                               be1, xbuf, xbf);
  k_ffn1 <<<(B_ * L_ / 16) * (FF_ / 16 / 8), 256, 0, stream>>>(xbf, w1bf, b1,
                                                               hbf);
  k_ffn2 <<<(B_ * L_ / 16) * (D_ / 16 / 8), 256, 0, stream>>>(hbf, w2bf, b2,
                                                              xbuf, ybuf);
  k_ln2  <<<B_ * L_, 256, 0, stream>>>(ybuf, g2, be2, out);
}
